// TransformerDecoderLayer_5308579577951
// MI455X (gfx1250) — compile-verified, matrix operand order checked by reading, not run on hardware
//
#include <hip/hip_runtime.h>
#include <hip/hip_bf16.h>

// ---------------- problem constants ----------------
#define DD   1024            // d_model
#define HH   16              // heads
#define HD   64              // head dim
#define FFD  2048            // ffn dim
#define EE   4               // experts
#define SS   1024            // seq
#define BB   8               // batch
#define NTOK (SS*BB)         // 8192 tokens
#define D3   (3*DD)          // 3072
#define BROW 40              // padded LDS row stride for B tiles (elements)

typedef __bf16 v16bf __attribute__((ext_vector_type(16)));
typedef float  v8f   __attribute__((ext_vector_type(8)));

static __device__ inline v8f vzero8() {
  v8f z;
#pragma unroll
  for (int i = 0; i < 8; ++i) z[i] = 0.f;
  return z;
}

// WMMA: D = A(16x32 bf16) * B(32x16 bf16) + C(16x16 f32)
static __device__ inline v8f bwmma(v16bf a, v16bf b, v8f c) {
  return __builtin_amdgcn_wmma_f32_16x16x32_bf16(false, a, false, b, (short)0, c,
                                                 false, false);
}

// Load one lane's A-or-B fragment from row-major storage.
// p points at element (row, k0 + kbase); elements e<8 -> K = kbase+e,
// e>=8 -> K = kbase+16+(e-8)   (ISA 7.12.2 16-bit A 16x32 layout).
static __device__ inline v16bf frag_ld(const __bf16* p) {
  v16bf f;
#pragma unroll
  for (int e = 0; e < 8; ++e) f[e] = p[e];
#pragma unroll
  for (int e = 0; e < 8; ++e) f[8 + e] = p[16 + e];
  return f;
}

// issue one async 16-byte global->LDS copy (ASYNCcnt-tracked)
static __device__ inline void async_cp16(unsigned lds_byte_addr, const void* gaddr) {
  asm volatile("global_load_async_to_lds_b128 %0, %1, off"
               :: "v"(lds_byte_addr), "v"(gaddr) : "memory");
}

// ---------------- elementwise helpers ----------------
__global__ void k_f32_to_bf16(const float* __restrict__ s, __bf16* __restrict__ d, int n) {
  int i = blockIdx.x * blockDim.x + threadIdx.x;
  if (i < n) d[i] = (__bf16)s[i];
}

__global__ void k_zero(float* __restrict__ p, int n) {
  int i = blockIdx.x * blockDim.x + threadIdx.x;
  if (i < n) p[i] = 0.f;
}

// ---------------- gating ----------------
__global__ void k_gate(const float* __restrict__ x, const float* __restrict__ Wg,
                       float* __restrict__ wexp, float* __restrict__ gsums) {
  int n = blockIdx.x * blockDim.x + threadIdx.x;
  if (n >= NTOK) return;
  const float* xr = x + (size_t)n * DD;
  float lg[4] = {0.f, 0.f, 0.f, 0.f};
  for (int d = 0; d < DD; ++d) {
    float xv = xr[d];
    const float* w = Wg + (size_t)d * EE;
#pragma unroll
    for (int e = 0; e < EE; ++e) lg[e] += xv * w[e];
  }
  float mx = lg[0];
#pragma unroll
  for (int e = 1; e < EE; ++e) mx = fmaxf(mx, lg[e]);
  float p[4], ssum = 0.f;
#pragma unroll
  for (int e = 0; e < EE; ++e) { p[e] = __expf(lg[e] - mx); ssum += p[e]; }
  float inv = 1.f / ssum;
#pragma unroll
  for (int e = 0; e < EE; ++e) p[e] *= inv;
  int imax = 0;
#pragma unroll
  for (int e = 1; e < EE; ++e) if (p[e] > p[imax]) imax = e;   // first-max like argmax
#pragma unroll
  for (int e = 0; e < EE; ++e) wexp[(size_t)e * NTOK + n] = (e == imax) ? p[e] : 0.f;
#pragma unroll
  for (int e = 0; e < EE; ++e) atomicAdd(&gsums[e], p[e]);
  atomicAdd(&gsums[4 + imax], 1.0f);
}

__global__ void k_aux(const float* __restrict__ gsums, float* __restrict__ out) {
  if (threadIdx.x == 0 && blockIdx.x == 0) {
    float a = 0.f;
    const float invN = 1.0f / (float)NTOK;
#pragma unroll
    for (int e = 0; e < EE; ++e) a += (gsums[e] * invN) * (gsums[4 + e] * invN);
    out[(size_t)NTOK * DD] = (float)EE * a;
  }
}

// ---------------- bf16 WMMA GEMM with async-LDS B staging ----------------
// C[m,n] = sum_k A[m,k]*W[n,k] + bias[n]   (A row-major bf16, W row-major [N,K] bf16)
// Block = 256 thr = 8 waves; tile 128(M) x 64(N); B tile double-buffered in LDS
// via GLOBAL_LOAD_ASYNC_TO_LDS_B128, overlapped with the WMMAs of the current tile.
__global__ void __launch_bounds__(256)
k_gemm(const __bf16* __restrict__ A, int lda,
       const __bf16* __restrict__ W, int ldw,
       const float* __restrict__ bias, const float* __restrict__ rowScale,
       float* __restrict__ Cf, __bf16* __restrict__ Cbf, int ldc,
       int K, int relu, int accum) {
  __shared__ __bf16 wtile[2][64 * BROW];

  int tid  = threadIdx.x;
  int lane = tid & 31;
  int wave = tid >> 5;
  int row0 = blockIdx.y * 128 + wave * 16;
  int col0 = blockIdx.x * 64;
  int r  = lane & 15;
  int kb = (lane >> 4) * 8;

  // async-copy mapping: thread t copies one 16B chunk; n = t>>2, chunk = t&3
  int cn  = tid >> 2;
  int cc4 = (tid & 3) * 8;
  const __bf16* wsrc = W + (size_t)(col0 + cn) * ldw + cc4;
  unsigned ldst0 = (unsigned)(unsigned long long)&wtile[0][cn * BROW + cc4];
  unsigned ldst1 = (unsigned)(unsigned long long)&wtile[1][cn * BROW + cc4];

  // prefetch k-tile 0 into buffer 0
  async_cp16(ldst0, wsrc);

  v8f acc[4];
#pragma unroll
  for (int j = 0; j < 4; ++j) acc[j] = vzero8();

  const __bf16* arow = A + (size_t)(row0 + r) * lda + kb;
  v16bf a_cur = frag_ld(arow);

  for (int k0 = 0; k0 < K; k0 += 32) {
    int cur = (k0 >> 5) & 1;
    // current tile resident (my wave's chunks), then all waves' chunks
    asm volatile("s_wait_asynccnt 0x0" ::: "memory");
    __syncthreads();
    // overlap: start DMA of next tile into the other buffer
    if (k0 + 32 < K) {
      async_cp16(cur ? ldst0 : ldst1, wsrc + k0 + 32);
    }
    // software-pipeline next A fragment
    v16bf a_nxt = a_cur;
    if (k0 + 32 < K) a_nxt = frag_ld(arow + k0 + 32);

    const __bf16* wb = &wtile[cur][0];
    v16bf b0 = frag_ld(wb + (0 * 16 + r) * BROW + kb);
    v16bf b1 = frag_ld(wb + (1 * 16 + r) * BROW + kb);
    v16bf b2 = frag_ld(wb + (2 * 16 + r) * BROW + kb);
    v16bf b3 = frag_ld(wb + (3 * 16 + r) * BROW + kb);
    acc[0] = bwmma(a_cur, b0, acc[0]);
    acc[1] = bwmma(a_cur, b1, acc[1]);
    acc[2] = bwmma(a_cur, b2, acc[2]);
    acc[3] = bwmma(a_cur, b3, acc[3]);
    a_cur = a_nxt;
  }

  int cr = (lane >> 4) * 8;   // row offset of acc element 0
  int cc = lane & 15;         // column within 16-tile
#pragma unroll
  for (int j = 0; j < 4; ++j) {
    int col = col0 + j * 16 + cc;
    float bs = bias ? bias[col] : 0.f;
#pragma unroll
    for (int v = 0; v < 8; ++v) {
      int mrow = row0 + cr + v;
      float val = acc[j][v] + bs;
      if (relu) val = fmaxf(val, 0.f);
      if (rowScale) val *= rowScale[mrow];
      size_t off = (size_t)mrow * ldc + col;
      if (accum) val += Cf[off];
      if (Cf)  Cf[off]  = val;
      if (Cbf) Cbf[off] = (__bf16)val;
    }
  }
}

// ---------------- flash attention (one wave / 16 query rows) ----------------
// qkv: bf16 [NTOK, 3072], row n = s*BB+b ; Q at h*64, K at 1024+h*64, V at 2048+h*64
// ctx: bf16 [NTOK, 1024]. V tiles (32x64) double-buffered in LDS via async DMA.
__global__ void __launch_bounds__(32)
k_attn(const __bf16* __restrict__ qkv, __bf16* __restrict__ ctx) {
  __shared__ __bf16 pbuf[16 * 32];
  __shared__ __bf16 vbuf[2][32 * 64];
  int lane = threadIdx.x;
  int bh = blockIdx.y;
  int b = bh / HH, h = bh % HH;
  int s0 = blockIdx.x * 16;
  int r  = lane & 15;
  int kb = (lane >> 4) * 8;

  const __bf16* kbase = qkv + (size_t)b * D3 + DD     + h * HD;
  const __bf16* vbase = qkv + (size_t)b * D3 + 2 * DD + h * HD;
  const size_t tstride = (size_t)BB * D3;
  const float sc = 0.125f;   // 64^-0.5

  // Q fragments for d = 0..31 and 32..63
  const __bf16* qrow = qkv + (size_t)((s0 + r) * BB + b) * D3 + h * HD;
  v16bf aq0 = frag_ld(qrow + kb);
  v16bf aq1 = frag_ld(qrow + 32 + kb);

  float m[8], l[8];
#pragma unroll
  for (int v = 0; v < 8; ++v) { m[v] = -1e30f; l[v] = 0.f; }
  v8f o[4];
#pragma unroll
  for (int j = 0; j < 4; ++j) o[j] = vzero8();

  // async prefetch of a 32x64 V tile: 256 chunks of 16B, 8 per lane
  auto vprefetch = [&](int t0, int bufi) {
#pragma unroll
    for (int c = 0; c < 8; ++c) {
      int chunk = c * 32 + lane;
      int t   = chunk >> 3;
      int cc8 = (chunk & 7) * 8;
      const __bf16* g = vbase + (size_t)(t0 + t) * tstride + cc8;
      unsigned ldsb = (unsigned)(unsigned long long)&vbuf[bufi][t * 64 + cc8];
      async_cp16(ldsb, g);
    }
  };
  vprefetch(0, 0);

  for (int t0 = 0; t0 < SS; t0 += 32) {
    int cur = (t0 >> 5) & 1;
    // ---- scores: two 16x16 tiles over columns t0..t0+15 / +16..+31 ----
    v8f sA = vzero8(), sB = vzero8();
    {
      const __bf16* kr0 = kbase + (size_t)(t0 + r)      * tstride + kb;
      const __bf16* kr1 = kbase + (size_t)(t0 + 16 + r) * tstride + kb;
      v16bf b00 = frag_ld(kr0);        // d 0..31
      v16bf b01 = frag_ld(kr0 + 32);   // d 32..63
      v16bf b10 = frag_ld(kr1);
      v16bf b11 = frag_ld(kr1 + 32);
      sA = bwmma(aq0, b00, sA);  sA = bwmma(aq1, b01, sA);
      sB = bwmma(aq0, b10, sB);  sB = bwmma(aq1, b11, sB);
    }
    // ---- online softmax over the 32 new columns ----
    float esc[8];
#pragma unroll
    for (int v = 0; v < 8; ++v) {
      float sa = sA[v] * sc, sb = sB[v] * sc;
      float mx = fmaxf(sa, sb);
      mx = fmaxf(mx, __shfl_xor(mx, 1, 32));
      mx = fmaxf(mx, __shfl_xor(mx, 2, 32));
      mx = fmaxf(mx, __shfl_xor(mx, 4, 32));
      mx = fmaxf(mx, __shfl_xor(mx, 8, 32));
      float nm = fmaxf(m[v], mx);
      float ee = __expf(m[v] - nm);
      float pa = __expf(sa - nm);
      float pb = __expf(sb - nm);
      float rs = pa + pb;
      rs += __shfl_xor(rs, 1, 32);
      rs += __shfl_xor(rs, 2, 32);
      rs += __shfl_xor(rs, 4, 32);
      rs += __shfl_xor(rs, 8, 32);
      l[v] = l[v] * ee + rs;
      m[v] = nm;
      esc[v] = ee;
      sA[v] = pa;  sB[v] = pb;          // reuse accumulators as P
    }
#pragma unroll
    for (int j = 0; j < 4; ++j)
#pragma unroll
      for (int v = 0; v < 8; ++v) o[j][v] *= esc[v];

    // ---- re-fragment P: C-layout regs -> LDS -> A-layout frag ----
#pragma unroll
    for (int v = 0; v < 8; ++v) {
      int prow = (lane >> 4) * 8 + v;
      pbuf[prow * 32 + r]      = (__bf16)sA[v];
      pbuf[prow * 32 + 16 + r] = (__bf16)sB[v];
    }
    v16bf ap = frag_ld(&pbuf[r * 32 + kb]);  // LDS in-order within a wave

    // ---- V tile pipeline: start next DMA, wait for current ----
    if (t0 + 32 < SS) {
      // prior LDS reads of the buffer we are about to overwrite are done
      asm volatile("s_wait_dscnt 0x0" ::: "memory");
      vprefetch(t0 + 32, cur ^ 1);
      // 8 newest asyncs may be pending; first 8 (current tile) must be done
      asm volatile("s_wait_asynccnt 0x8" ::: "memory");
    } else {
      asm volatile("s_wait_asynccnt 0x0" ::: "memory");
    }

    // ---- O += P @ V  (B fragments gathered from LDS) ----
    const __bf16* vt = &vbuf[cur][0];
#pragma unroll
    for (int j = 0; j < 4; ++j) {
      int dcol = j * 16 + r;
      v16bf bv;
#pragma unroll
      for (int e = 0; e < 8; ++e) bv[e]     = vt[(kb + e)      * 64 + dcol];
#pragma unroll
      for (int e = 0; e < 8; ++e) bv[8 + e] = vt[(kb + 16 + e) * 64 + dcol];
      o[j] = bwmma(ap, bv, o[j]);
    }
  }

  // ---- normalize and write context ----
#pragma unroll
  for (int v = 0; v < 8; ++v) {
    int srow = s0 + (lane >> 4) * 8 + v;
    float inv = 1.f / l[v];
#pragma unroll
    for (int j = 0; j < 4; ++j)
      ctx[(size_t)(srow * BB + b) * DD + h * HD + j * 16 + r] = (__bf16)(o[j][v] * inv);
  }
}

// ---------------- layer norms ----------------
static __device__ inline float blk_sum(float v) {
  __shared__ float sb[256];
  int t = threadIdx.x;
  sb[t] = v;
  __syncthreads();
  for (int s = 128; s > 0; s >>= 1) {
    if (t < s) sb[t] += sb[t + s];
    __syncthreads();
  }
  float r = sb[0];
  __syncthreads();
  return r;
}

// x = LN2(LN1(a + b)) ; writes f32 and bf16
__global__ void __launch_bounds__(256)
k_ln2(const float* __restrict__ a, const float* __restrict__ b,
      const float* __restrict__ g1, const float* __restrict__ be1,
      const float* __restrict__ g2, const float* __restrict__ be2,
      float* __restrict__ outF, __bf16* __restrict__ outBf) {
  int row = blockIdx.x, t = threadIdx.x;
  const float* ar = a + (size_t)row * DD;
  const float* br = b + (size_t)row * DD;
  float x[4];
#pragma unroll
  for (int i = 0; i < 4; ++i) x[i] = ar[t + i * 256] + br[t + i * 256];

  float s = 0.f;
#pragma unroll
  for (int i = 0; i < 4; ++i) s += x[i];
  float mean = blk_sum(s) * (1.f / DD);
  float vs = 0.f;
#pragma unroll
  for (int i = 0; i < 4; ++i) { float c = x[i] - mean; vs += c * c; }
  float inv = rsqrtf(blk_sum(vs) * (1.f / DD) + 1e-5f);
#pragma unroll
  for (int i = 0; i < 4; ++i) {
    int d = t + i * 256;
    x[i] = (x[i] - mean) * inv * g1[d] + be1[d];
  }
  // second norm
  s = 0.f;
#pragma unroll
  for (int i = 0; i < 4; ++i) s += x[i];
  mean = blk_sum(s) * (1.f / DD);
  vs = 0.f;
#pragma unroll
  for (int i = 0; i < 4; ++i) { float c = x[i] - mean; vs += c * c; }
  inv = rsqrtf(blk_sum(vs) * (1.f / DD) + 1e-5f);
#pragma unroll
  for (int i = 0; i < 4; ++i) {
    int d = t + i * 256;
    float y = (x[i] - mean) * inv * g2[d] + be2[d];
    outF[(size_t)row * DD + d]  = y;
    outBf[(size_t)row * DD + d] = (__bf16)y;
  }
}

// out = LN(a + b)
__global__ void __launch_bounds__(256)
k_ln(const float* __restrict__ a, const float* __restrict__ b,
     const float* __restrict__ g, const float* __restrict__ be,
     float* __restrict__ outF) {
  int row = blockIdx.x, t = threadIdx.x;
  const float* ar = a + (size_t)row * DD;
  const float* br = b + (size_t)row * DD;
  float x[4];
#pragma unroll
  for (int i = 0; i < 4; ++i) x[i] = ar[t + i * 256] + br[t + i * 256];
  float s = 0.f;
#pragma unroll
  for (int i = 0; i < 4; ++i) s += x[i];
  float mean = blk_sum(s) * (1.f / DD);
  float vs = 0.f;
#pragma unroll
  for (int i = 0; i < 4; ++i) { float c = x[i] - mean; vs += c * c; }
  float inv = rsqrtf(blk_sum(vs) * (1.f / DD) + 1e-5f);
#pragma unroll
  for (int i = 0; i < 4; ++i) {
    int d = t + i * 256;
    outF[(size_t)row * DD + d] = (x[i] - mean) * inv * g[d] + be[d];
  }
}

// ---------------- launch ----------------
extern "C" void kernel_launch(void* const* d_in, const int* in_sizes, int n_in,
                              void* d_out, int out_size, void* d_ws, size_t ws_size,
                              hipStream_t stream) {
  (void)in_sizes; (void)n_in; (void)out_size; (void)ws_size;
  const float* tgt  = (const float*)d_in[0];
  const float* Wg   = (const float*)d_in[1];
  const float* Wqkv = (const float*)d_in[2];
  const float* bqkv = (const float*)d_in[3];
  const float* Wo   = (const float*)d_in[4];
  const float* bo   = (const float*)d_in[5];
  const float* W1   = (const float*)d_in[6];
  const float* b1   = (const float*)d_in[7];
  const float* W2   = (const float*)d_in[8];
  const float* b2   = (const float*)d_in[9];
  const float* g1   = (const float*)d_in[10];
  const float* be1  = (const float*)d_in[11];
  const float* g2   = (const float*)d_in[12];
  const float* be2  = (const float*)d_in[13];
  const float* g3   = (const float*)d_in[14];
  const float* be3  = (const float*)d_in[15];
  float* out = (float*)d_out;

  char* ws = (char*)d_ws;
  size_t off = 0;
  auto alloc = [&](size_t bytes) -> char* {
    char* p = ws + off;
    off = (off + bytes + 255) & ~(size_t)255;
    return p;
  };

  __bf16* Xbf    = (__bf16*)alloc((size_t)NTOK * DD * 2);
  __bf16* Wqkvbf = (__bf16*)alloc((size_t)EE * D3 * DD * 2);
  __bf16* Wobf   = (__bf16*)alloc((size_t)EE * DD * DD * 2);
  __bf16* W1bf   = (__bf16*)alloc((size_t)FFD * DD * 2);
  __bf16* W2bf   = (__bf16*)alloc((size_t)DD * FFD * 2);
  __bf16* qkvbf  = (__bf16*)alloc((size_t)NTOK * D3 * 2);   // reused as h1 (bf16)
  __bf16* ctxbf  = (__bf16*)alloc((size_t)NTOK * DD * 2);
  float*  attnO  = (float*) alloc((size_t)NTOK * DD * 4);   // reused as ff2
  float*  x2f    = (float*) alloc((size_t)NTOK * DD * 4);
  __bf16* xbf    = (__bf16*)alloc((size_t)NTOK * DD * 2);
  float*  wexp   = (float*) alloc((size_t)EE * NTOK * 4);
  float*  gsums  = (float*) alloc(256);

  auto cvt = [&](const float* s, __bf16* d, size_t n) {
    k_f32_to_bf16<<<dim3((unsigned)((n + 255) / 256)), 256, 0, stream>>>(s, d, (int)n);
  };
  cvt(tgt,  Xbf,    (size_t)NTOK * DD);
  cvt(Wqkv, Wqkvbf, (size_t)EE * D3 * DD);
  cvt(Wo,   Wobf,   (size_t)EE * DD * DD);
  cvt(W1,   W1bf,   (size_t)FFD * DD);
  cvt(W2,   W2bf,   (size_t)DD * FFD);

  k_zero<<<dim3((NTOK * DD + 255) / 256), 256, 0, stream>>>(attnO, NTOK * DD);
  k_zero<<<1, 256, 0, stream>>>(gsums, 8);

  k_gate<<<dim3(NTOK / 256), 256, 0, stream>>>(tgt, Wg, wexp, gsums);

  for (int e = 0; e < EE; ++e) {
    // qkv = X @ Wqkv[e]^T + bqkv[e]   -> bf16
    k_gemm<<<dim3(D3 / 64, NTOK / 128), 256, 0, stream>>>(
        Xbf, DD, Wqkvbf + (size_t)e * D3 * DD, DD,
        bqkv + (size_t)e * D3, nullptr, nullptr, qkvbf, D3, DD, 0, 0);
    // flash attention per (b,h, 16-row q tile)
    k_attn<<<dim3(SS / 16, BB * HH), 32, 0, stream>>>(qkvbf, ctxbf);
    // attnO += gate_e(row) * (ctx @ Wo[e]^T + bo[e])
    k_gemm<<<dim3(DD / 64, NTOK / 128), 256, 0, stream>>>(
        ctxbf, DD, Wobf + (size_t)e * DD * DD, DD,
        bo + (size_t)e * DD, wexp + (size_t)e * NTOK, attnO, nullptr, DD, DD, 0, 1);
  }

  // x = LN2(LN1(tgt + attnO))
  k_ln2<<<dim3(NTOK), 256, 0, stream>>>(tgt, attnO, g1, be1, g2, be2, x2f, xbf);

  // FFN
  __bf16* h1bf = qkvbf;
  k_gemm<<<dim3(FFD / 64, NTOK / 128), 256, 0, stream>>>(
      xbf, DD, W1bf, DD, b1, nullptr, nullptr, h1bf, FFD, DD, 1, 0);
  float* ff2 = attnO;
  k_gemm<<<dim3(DD / 64, NTOK / 128), 256, 0, stream>>>(
      h1bf, FFD, W2bf, FFD, b2, nullptr, ff2, nullptr, DD, FFD, 0, 0);

  // out = LN(x + ff2) ; aux scalar
  k_ln<<<dim3(NTOK), 256, 0, stream>>>(x2f, ff2, g3, be3, out);
  k_aux<<<1, 1, 0, stream>>>(gsums, out);
}
